// DyConv_63445256897038
// MI455X (gfx1250) — matrix operand order repaired and checked by reading, not compiled, not run on hardware
//
#include <hip/hip_runtime.h>
#include <hip/hip_bf16.h>
#include <math.h>

// Shapes from reference: B=16, C=256, H=W=64 (HW=4096), K=4, RED=4 (C/RED=64)
#define BATCH 16
#define CCH   256
#define HWSZ  4096
#define NEXP  4

typedef __attribute__((ext_vector_type(16))) __bf16         v16bf;
typedef __attribute__((ext_vector_type(8)))  float          v8f;
typedef __attribute__((ext_vector_type(8)))  unsigned short v8u;
typedef __attribute__((ext_vector_type(16))) unsigned short v16u;

static __device__ __forceinline__ unsigned short f2bf_bits(float f) {
    unsigned u = __builtin_bit_cast(unsigned, f);
    u += 0x7FFFu + ((u >> 16) & 1u);          // round-to-nearest-even
    return (unsigned short)(u >> 16);
}

// ---------------------------------------------------------------------------
// Kernel 1: pooled[b][c] = mean over HW.  One wave32 per (b,c) pair.
// ---------------------------------------------------------------------------
__global__ __launch_bounds__(256) void pool_kernel(const float* __restrict__ x,
                                                   float* __restrict__ pooled) {
    const int wave = threadIdx.x >> 5;
    const int lane = threadIdx.x & 31;
    const int p    = blockIdx.x * 8 + wave;              // 0 .. B*C-1
    const float4* src4 = reinterpret_cast<const float4*>(x + (size_t)p * HWSZ);
    float s = 0.f;
#pragma unroll 4
    for (int i = 0; i < 32; ++i) {                       // 32 * 4 * 32 lanes = 4096
        float4 v = src4[lane + i * 32];
        s += v.x + v.y + v.z + v.w;
    }
#pragma unroll
    for (int off = 16; off; off >>= 1) s += __shfl_xor(s, off, 32);
    if (lane == 0) pooled[p] = s * (1.0f / (float)HWSZ);
}

// ---------------------------------------------------------------------------
// Kernel 2: gating MLP + softmax.  Single block, tiny work.
// hidden = relu(pooled @ w1^T) : (16,64);  alpha = softmax(hidden @ w2^T) : (16,4)
// ---------------------------------------------------------------------------
__global__ __launch_bounds__(1024) void gate_kernel(const float* __restrict__ pooled,
                                                    const float* __restrict__ w1,
                                                    const float* __restrict__ w2,
                                                    float* __restrict__ alpha) {
    __shared__ float sp[BATCH * CCH];     // pooled
    __shared__ float sh[BATCH * 64];      // hidden
    __shared__ float sl[BATCH * NEXP];    // logits
    const int t = threadIdx.x;
    for (int i = t; i < BATCH * CCH; i += 1024) sp[i] = pooled[i];
    __syncthreads();

    const int b = t >> 6, j = t & 63;     // one thread per hidden element
    {
        const float* pr = sp + b * CCH;
        const float* wr = w1 + j * CCH;
        float acc = 0.f;
#pragma unroll 8
        for (int c = 0; c < CCH; ++c) acc += pr[c] * wr[c];
        sh[b * 64 + j] = fmaxf(acc, 0.f);
    }
    __syncthreads();

    if (t < BATCH * NEXP) {
        const int bb = t >> 2, k = t & 3;
        const float* hr  = sh + bb * 64;
        const float* w2r = w2 + k * 64;
        float acc = 0.f;
#pragma unroll 8
        for (int j2 = 0; j2 < 64; ++j2) acc += hr[j2] * w2r[j2];
        sl[t] = acc;
    }
    __syncthreads();

    if (t < BATCH) {
        float l0 = sl[t*4+0], l1 = sl[t*4+1], l2 = sl[t*4+2], l3 = sl[t*4+3];
        float m  = fmaxf(fmaxf(l0, l1), fmaxf(l2, l3));
        float e0 = expf(l0 - m), e1 = expf(l1 - m), e2 = expf(l2 - m), e3 = expf(l3 - m);
        float inv = 1.f / (e0 + e1 + e2 + e3);
        alpha[t*4+0] = e0*inv; alpha[t*4+1] = e1*inv;
        alpha[t*4+2] = e2*inv; alpha[t*4+3] = e3*inv;
    }
}

// ---------------------------------------------------------------------------
// Kernel 3: w_mixed[b][o][c] = sum_k alpha[b][k] * experts[k][o][c], stored bf16.
// 16 contiguous elements per thread; 32B-aligned v16u store.
// ---------------------------------------------------------------------------
__global__ __launch_bounds__(256) void mix_kernel(const float* __restrict__ experts,
                                                  const float* __restrict__ alpha,
                                                  unsigned short* __restrict__ wmix) {
    const size_t e0 = ((size_t)blockIdx.x * blockDim.x + threadIdx.x) * 16;
    const int    b  = (int)(e0 >> 16);                   // 65536 = C*C per batch
    const size_t oc = e0 & 65535;
    const float a0 = alpha[b*4+0], a1 = alpha[b*4+1];
    const float a2 = alpha[b*4+2], a3 = alpha[b*4+3];
    const float* ep = experts + oc;
    v16u ov;
#pragma unroll
    for (int q = 0; q < 4; ++q) {
        float4 v0 = *(const float4*)(ep + 0u*65536u + q*4);
        float4 v1 = *(const float4*)(ep + 1u*65536u + q*4);
        float4 v2 = *(const float4*)(ep + 2u*65536u + q*4);
        float4 v3 = *(const float4*)(ep + 3u*65536u + q*4);
        ov[q*4+0] = f2bf_bits(a0*v0.x + a1*v1.x + a2*v2.x + a3*v3.x);
        ov[q*4+1] = f2bf_bits(a0*v0.y + a1*v1.y + a2*v2.y + a3*v3.y);
        ov[q*4+2] = f2bf_bits(a0*v0.z + a1*v1.z + a2*v2.z + a3*v3.z);
        ov[q*4+3] = f2bf_bits(a0*v0.w + a1*v1.w + a2*v2.w + a3*v3.w);
    }
    *(v16u*)(wmix + e0) = ov;
}

// ---------------------------------------------------------------------------
// Kernel 4: out[b] = w_mixed[b] (256x256) @ x[b] (256x4096) via bf16 WMMA.
// Block: 512 thr (16 waves), tile = full M=256 x N=64, full K=256 in LDS.
// LDS holds the x tile transposed: xb[n][c] (bf16), row stride 264 so each
// lane's B fragment (16 contiguous K) is 16-byte aligned and bank-spread.
// ---------------------------------------------------------------------------
#define NT       64
#define LDSTRIDE 264

__global__ __launch_bounds__(512) void gemm_kernel(const float* __restrict__ x,
                                                   const unsigned short* __restrict__ wmix,
                                                   float* __restrict__ out) {
    __shared__ __align__(16) unsigned short xb[NT * LDSTRIDE];   // ~33 KB
    const int b   = blockIdx.y;
    const int hw0 = blockIdx.x * NT;
    const float* xsrc = x + (size_t)b * CCH * HWSZ;

    // Fill LDS: x[b][c][hw0+n] -> xb[n][c] as bf16 (coalesced global reads over n)
    {
        const int n  = threadIdx.x & 63;
        const int c0 = threadIdx.x >> 6;                 // 0..7
#pragma unroll 4
        for (int i = 0; i < 32; ++i) {
            const int c = c0 + i * 8;
            xb[n * LDSTRIDE + c] = f2bf_bits(xsrc[(size_t)c * HWSZ + hw0 + n]);
        }
    }
    __syncthreads();

    const int wave  = threadIdx.x >> 5;
    const int lane  = threadIdx.x & 31;
    const int half  = lane >> 4;                         // 0 or 1
    const int lan16 = lane & 15;
    const int m0    = wave * 16;                         // this wave's M tile

    // A row base for this lane (row-major w_mixed[b], 256 cols)
    const unsigned short* A = wmix + ((size_t)b * 256 + m0 + lan16) * 256;

    v8f acc[4] = {};                                     // 4 N-subtiles of 16

#pragma unroll
    for (int ks = 0; ks < 8; ++ks) {                     // K = 8 * 32
        const int k0 = ks * 32;
        const int kb = k0 + half * 16;
        // A fragment: lane<16 -> K {0..15}; lane>=16 -> K {16..31}
        v8u alo = *(const v8u*)(A + kb);
        v8u ahi = *(const v8u*)(A + kb + 8);
        v16bf af = __builtin_bit_cast(v16bf,
            __builtin_shufflevector(alo, ahi, 0,1,2,3,4,5,6,7,8,9,10,11,12,13,14,15));
        if (ks < 7) __builtin_prefetch(A + kb + 32, 0, 1);
#pragma unroll
        for (int nt = 0; nt < 4; ++nt) {
            // B fragment: col = nt*16 + lan16, K = k0 + half*16 .. +15 (contiguous)
            const unsigned short* bp = &xb[(nt*16 + lan16) * LDSTRIDE + k0 + half*16];
            v8u blo = *(const v8u*)bp;
            v8u bhi = *(const v8u*)(bp + 8);
            v16bf bf = __builtin_bit_cast(v16bf,
                __builtin_shufflevector(blo, bhi, 0,1,2,3,4,5,6,7,8,9,10,11,12,13,14,15));
            acc[nt] = __builtin_amdgcn_wmma_f32_16x16x32_bf16(
                false, af, false, bf, (short)0, acc[nt], false, false);
        }
    }

    // Store D: lane<16 -> rows m0+0..7, lane>=16 -> rows m0+8..15; col = lan16
    float* ob = out + ((size_t)b * 256 + m0 + half * 8) * HWSZ + hw0 + lan16;
#pragma unroll
    for (int nt = 0; nt < 4; ++nt) {
        float* oc = ob + nt * 16;
#pragma unroll
        for (int i = 0; i < 8; ++i) oc[(size_t)i * HWSZ] = acc[nt][i];
    }
}

// ---------------------------------------------------------------------------
extern "C" void kernel_launch(void* const* d_in, const int* in_sizes, int n_in,
                              void* d_out, int out_size, void* d_ws, size_t ws_size,
                              hipStream_t stream) {
    const float* x       = (const float*)d_in[0];   // (16,256,64,64)
    const float* experts = (const float*)d_in[1];   // (4,256,256,1,1)
    const float* w1      = (const float*)d_in[2];   // (64,256)
    const float* w2      = (const float*)d_in[3];   // (4,64)
    float* out = (float*)d_out;                     // (16,256,64,64)

    char* ws = (char*)d_ws;
    float*          pooled = (float*)(ws);                 // 16KB
    float*          alpha  = (float*)(ws + 16384);         // 256B
    unsigned short* wmix   = (unsigned short*)(ws + 32768);// 2MB bf16

    pool_kernel<<<(BATCH * CCH) / 8, 256, 0, stream>>>(x, pooled);
    gate_kernel<<<1, 1024, 0, stream>>>(pooled, w1, w2, alpha);
    mix_kernel<<<(BATCH * CCH * CCH) / (256 * 16), 256, 0, stream>>>(experts, alpha, wmix);
    gemm_kernel<<<dim3(HWSZ / NT, BATCH), 512, 0, stream>>>(x, wmix, out);
}
